// BackProjectionNet_30167850287302
// MI455X (gfx1250) — matrix operand order repaired and checked by reading, not compiled, not run on hardware
//
#include <hip/hip_runtime.h>

typedef __attribute__((ext_vector_type(16))) _Float16 v16h;
typedef __attribute__((ext_vector_type(8)))  _Float16 v8h;
typedef __attribute__((ext_vector_type(8)))  float    v8f;
typedef __attribute__((ext_vector_type(4)))  unsigned int u32x4;
typedef __attribute__((ext_vector_type(8)))  int i32x8;
typedef __attribute__((ext_vector_type(4)))  int i32x4;

#define N_ANGLES 120
#define NPAIRS   60           // two angles per WMMA K=32
#define LRR      128
#define LZZ      192
#define PADP     27
#define LPP      182          // LRR + 2*PADP
#define WSW      216          // ws width in columns; cc = padded_col + 1, zero-padded
#define CCOFF    28           // image col j = cc - CCOFF
#define TILE_Y   8            // WG output tile: 8 y rows x 8 x cols
#define TILE_X   8
#define KWIN     16           // per-angle source-column window
#define LDS_STRIDE 40         // f16 per z-row: 16+16 window cols + 8 pad (TDM pad)
#define NZT      12           // 192 / 16 z-tiles
#define NBUF     4            // LDS buffers: one barrier/iter, TDM 2 pairs ahead

// ---------------- prologue: f32 image -> padded, transposed f16 in ws ----------------
// ws layout: [angle][z][WSW columns] f16, columns contiguous, zero outside image.
__global__ __launch_bounds__(256)
void cvt_kernel(const float* __restrict__ img, _Float16* __restrict__ ws)
{
    const size_t e = (size_t)blockIdx.x * 256 + threadIdx.x;
    if (e >= (size_t)N_ANGLES * LZZ * WSW) return;
    const int a  = (int)(e / (LZZ * WSW));
    const int r  = (int)(e - (size_t)a * (LZZ * WSW));
    const int z  = r / WSW;
    const int cc = r - z * WSW;
    const int j  = cc - CCOFF;
    const float v = (j >= 0 && j < LRR) ? img[((size_t)a * LRR + j) * LZZ + z] : 0.f;
    ws[e] = (_Float16)v;
}

// ---------------- helpers ----------------
typedef __attribute__((address_space(3))) const void* lds_cptr_t;
static __device__ __forceinline__ unsigned lds_off_of(const void* p) {
    return (unsigned)(uintptr_t)(lds_cptr_t)p;   // low 32 bits of shared ptr = LDS byte addr
}

static __device__ __forceinline__ void angle_trig(int a, float& sp, float& cp) {
    const float phi = 4.71238898038468986f
                    - (float)a * (6.28318530717958648f / (float)N_ANGLES);
    __sincosf(phi, &sp, &cp);
}

// ws column of the window start (= floor over the WG tile of sx, +1 for ws pad offset)
static __device__ __forceinline__ int window_cc(float sp, float cp, float ycA, float ycB,
                                                float xcA, float xcB, float c) {
    const float sxmin = fminf(sp * ycA, sp * ycB) + fminf(cp * xcA, cp * xcB) + c;
    return (int)floorf(sxmin) + 1;
}

// One TDM load: 16 cols x 192 z f16 tile -> LDS rows of stride 80B (32B data + 48B skip).
static __device__ __forceinline__ void tdm_issue16(const _Float16* wsimg, int a, int cc,
                                                   unsigned lds_off)
{
    const unsigned long long g =
        (unsigned long long)(uintptr_t)(wsimg + ((size_t)a * LZZ) * WSW + cc);

    u32x4 g0;
    g0[0] = 1u;                                       // count=1, user descriptor
    g0[1] = lds_off;                                  // lds_addr (bytes)
    g0[2] = (unsigned)(g & 0xFFFFFFFFu);              // global_addr lo
    g0[3] = (unsigned)((g >> 32) & 0x1FFFFFFu)        // global_addr hi (bits 56:32)
          | (2u << 30);                               // type = 2 ("image")

    i32x8 g1;
    g1[0] = (1 << 16)                                 // data_size = 2 bytes
          | (1 << 20)                                 // pad_enable
          | (2 << 22)                                 // pad_interval: 8 DWORDs (32B)
          | (11 << 25);                               // pad_amount: 12 DWORDs (48B)
    g1[1] = 0;                                        // tensor_dim0 lo16 (1<<30 -> 0)
    g1[2] = 0x4000;                                   // tensor_dim0 hi16 (no OOB)
    g1[3] = 0x4000 | (KWIN << 16);                    // tensor_dim1 hi16, tile_dim0 = 16
    g1[4] = LZZ;                                      // tile_dim1 = 192, tile_dim2 = 0
    g1[5] = WSW;                                      // tensor_dim0_stride = 216
    g1[6] = 0;
    g1[7] = 0;

    const i32x4 gz = {0, 0, 0, 0};
#if defined(__clang_major__) && (__clang_major__ >= 23)
    const i32x8 gz8 = {0, 0, 0, 0, 0, 0, 0, 0};
    __builtin_amdgcn_tensor_load_to_lds(g0, g1, gz, gz, gz8, 0);
#else
    __builtin_amdgcn_tensor_load_to_lds(g0, g1, gz, gz, 0);
#endif
}

// Issue both tiles (two angles) of one pair into buffer `b`.
static __device__ __forceinline__ void issue_pair(const _Float16* wsimg, int pair,
                                                  unsigned ldsBase,
                                                  float ycA, float ycB,
                                                  float xcA, float xcB, float c)
{
    float sp, cp; int cc;
    angle_trig(2 * pair, sp, cp);
    cc = __builtin_amdgcn_readfirstlane(window_cc(sp, cp, ycA, ycB, xcA, xcB, c));
    tdm_issue16(wsimg, 2 * pair, cc, ldsBase);
    angle_trig(2 * pair + 1, sp, cp);
    cc = __builtin_amdgcn_readfirstlane(window_cc(sp, cp, ycA, ycB, xcA, xcB, c));
    tdm_issue16(wsimg, 2 * pair + 1, cc, ldsBase + 32);
}

// Bilinear weights for one output pixel at one angle; k0 = window column of w0.
static __device__ __forceinline__ void weights(float sp, float cp, float yc, float xc,
                                               float c, int cc,
                                               float& w0, float& w1, int& k0)
{
    const float sy = cp * yc - sp * xc + c;
    const float sx = sp * yc + cp * xc + c;
    const float fy0 = floorf(sy), fx0 = floorf(sx);
    const float wy = sy - fy0, wx = sx - fx0;
    const int y0 = (int)fy0, x0 = (int)fx0;
    const float vy0 = (y0     >= 0 && y0     < LPP) ? 1.f : 0.f;
    const float vy1 = (y0 + 1 >= 0 && y0 + 1 < LPP) ? 1.f : 0.f;
    const float vx0 = (x0     >= 0 && x0     < LPP) ? 1.f : 0.f;
    const float vx1 = (x0 + 1 >= 0 && x0 + 1 < LPP) ? 1.f : 0.f;
    const float fy = (1.f - wy) * vy0 + wy * vy1;
    w0 = (1.f - wx) * vx0 * fy;
    w1 = wx * vx1 * fy;
    k0 = (x0 + 1) - cc;                  // in [0,10] whenever weight nonzero
}

// ---------------- main kernel ----------------
// WG = 4 wave32; output tile 8y x 8x x 192z. Wave wv owns rows {2wv, 2wv+1}.
// WMMA M-tile = 2y x 8x (m = dy*8+dx), K = 32 packs TWO angles (16 cols each).
__global__ __launch_bounds__(128)
void bp_wmma_kernel(const _Float16* __restrict__ wsimg, float* __restrict__ out)
{
    const int xt   = blockIdx.x;             // 0..15 (8 x outputs each)
    const int yt   = blockIdx.y;             // 0..15 (8 y rows each)
    const int tid  = threadIdx.x;
    const int lane = tid & 31;
    const int wv   = tid >> 5;
    const int m    = lane & 15;              // M index: dy*8 + dx
    const bool hi  = lane >= 16;
    const int dy   = m >> 3;
    const int dx   = m & 7;

    __shared__ __align__(16) _Float16 shB[NBUF][LZZ * LDS_STRIDE];
    __shared__ float shNorm[4][16];

    const float c   = (LPP - 1) * 0.5f;      // 90.5
    const int   yb  = yt * TILE_Y;
    const int   xb  = xt * TILE_X;
    const float ycA = (float)(yb + PADP) - c;            // WG tile corners
    const float ycB = ycA + (float)(TILE_Y - 1);
    const float xcA = (float)(xb + PADP) - c;
    const float xcB = xcA + (float)(TILE_X - 1);
    const float yc  = (float)(yb + 2 * wv + dy + PADP) - c;  // this lane's output
    const float xc  = (float)(xb + dx + PADP) - c;

    v8f acc[NZT] = {};
    float norm = 0.f;
    const int koff = hi ? 16 : 0;

    // Pre-issue TDMs for pairs 0 and 1 (4 tiles in flight).
    if (wv == 0) {
        issue_pair(wsimg, 0, lds_off_of(&shB[0][0]), ycA, ycB, xcA, xcB, c);
        issue_pair(wsimg, 1, lds_off_of(&shB[1][0]), ycA, ycB, xcA, xcB, c);
    }

    for (int pr = 0; pr < NPAIRS; ++pr) {
        const int a0 = 2 * pr, a1 = a0 + 1;

        // ---- weights + A fragment for both angles (VALU, overlaps TDM) ----
        float sp0, cp0, sp1, cp1;
        angle_trig(a0, sp0, cp0);
        angle_trig(a1, sp1, cp1);
        const int cc0 = window_cc(sp0, cp0, ycA, ycB, xcA, xcB, c);
        const int cc1 = window_cc(sp1, cp1, ycA, ycB, xcA, xcB, c);

        float w0a, w1a, w0b, w1b; int ka, kb;
        weights(sp0, cp0, yc, xc, c, cc0, w0a, w1a, ka);
        weights(sp1, cp1, yc, xc, c, cc1, w0b, w1b, kb);
        norm += (w0a + w1a) + (w0b + w1b);

        const int ka1 = ka + 1;
        const int kb0 = kb + 16, kb1 = kb0 + 1;     // angle a1 lives in K 16..31
        const _Float16 ha0 = (_Float16)w0a, ha1 = (_Float16)w1a;
        const _Float16 hb0 = (_Float16)w0b, hb1 = (_Float16)w1b;

        v16h A;                                     // A 16x32 f16, contiguous K: k = i + 16*hi
        #pragma unroll
        for (int i = 0; i < 16; ++i) {
            const int kk = i + (hi ? 16 : 0);
            _Float16 v = (_Float16)0.f;
            v = (kk == ka ) ? ha0 : v;
            v = (kk == ka1) ? ha1 : v;
            v = (kk == kb0) ? hb0 : v;
            v = (kk == kb1) ? hb1 : v;
            A[i] = v;
        }

        // ---- pipeline: issue pair pr+2, complete pair pr (4-buffer rotation) ----
        // Safety: barrier (pr-1) completed => all waves finished reading pair pr-2,
        // whose buffer is exactly buf[(pr+2) & 3].
        if (wv == 0) {
            if (pr + 2 < NPAIRS) {
                issue_pair(wsimg, pr + 2, lds_off_of(&shB[(pr + 2) & (NBUF - 1)][0]),
                           ycA, ycB, xcA, xcB, c);
                __builtin_amdgcn_s_wait_tensorcnt(4);   // pairs pr+1, pr+2 outstanding
            } else if (pr + 1 < NPAIRS) {
                __builtin_amdgcn_s_wait_tensorcnt(2);   // pair pr+1 outstanding
            } else {
                __builtin_amdgcn_s_wait_tensorcnt(0);
            }
        }
        __syncthreads();                 // pair pr's tiles visible; sole barrier per iter

        // ---- 12 WMMAs, B loads pipelined TWO z-tiles ahead (dscnt <= 4 waits) ----
        const _Float16* __restrict__ buf = &shB[pr & (NBUF - 1)][0];
        v8h clo = *(const v8h*)&buf[(0 * 16 + m) * LDS_STRIDE + koff];
        v8h chi = *(const v8h*)&buf[(0 * 16 + m) * LDS_STRIDE + koff + 8];
        v8h nlo = *(const v8h*)&buf[(1 * 16 + m) * LDS_STRIDE + koff];
        v8h nhi = *(const v8h*)&buf[(1 * 16 + m) * LDS_STRIDE + koff + 8];
        #pragma unroll
        for (int t = 0; t < NZT; ++t) {
            const v8h flo = clo, fhi = chi;
            clo = nlo; chi = nhi;
            if (t + 2 < NZT) {
                const int zn = (t + 2) * 16 + m;
                nlo = *(const v8h*)&buf[zn * LDS_STRIDE + koff];
                nhi = *(const v8h*)&buf[zn * LDS_STRIDE + koff + 8];
            }
            const v16h B = __builtin_shufflevector(flo, fhi,
                             0,1,2,3,4,5,6,7,8,9,10,11,12,13,14,15);
            acc[t] = __builtin_amdgcn_wmma_f32_16x16x32_f16(
                         false, A, false, B, (short)0, acc[t], false, false);
        }
        // no trailing barrier: 4-buffer rotation + top barrier provide safety
    }

    // ---- normalization + store ----
    if (!hi) shNorm[wv][m] = norm;
    __syncthreads();

    #pragma unroll
    for (int r = 0; r < 8; ++r) {
        const int mm = r + (hi ? 8 : 0);         // C/D: VGPR r -> M = r + (hi?8:0)
        const int oy = yb + 2 * wv + (mm >> 3);
        const int ox = xb + (mm & 7);
        const float den = shNorm[wv][mm] + 1e-11f;
        float* __restrict__ o = out + ((size_t)oy * LRR + ox) * (size_t)LZZ;
        #pragma unroll
        for (int t = 0; t < NZT; ++t) {
            o[t * 16 + m] = acc[t][r] / den;     // N = lane&15 = m -> z = t*16+m
        }
    }
}

extern "C" void kernel_launch(void* const* d_in, const int* in_sizes, int n_in,
                              void* d_out, int out_size, void* d_ws, size_t ws_size,
                              hipStream_t stream)
{
    (void)in_sizes; (void)n_in; (void)out_size; (void)ws_size;
    const float* img = (const float*)d_in[0];    // [1,120,128,192] f32
    float* out = (float*)d_out;                  // [1,128,128,192] f32
    _Float16* ws = (_Float16*)d_ws;              // [120][192][216] f16 (~9.95 MB)

    const size_t nconv = (size_t)N_ANGLES * LZZ * WSW;
    hipLaunchKernelGGL(cvt_kernel, dim3((unsigned)((nconv + 255) / 256)), dim3(256),
                       0, stream, img, ws);

    hipLaunchKernelGGL(bp_wmma_kernel, dim3(LRR / TILE_X, LRR / TILE_Y), dim3(128),
                       0, stream, ws, out);
}
